// HeteroGNN_25520695673327
// MI455X (gfx1250) — matrix operand order repaired and checked by reading, not compiled, not run on hardware
//
#include <hip/hip_runtime.h>
#include <hip/hip_bf16.h>
#include <math.h>

#define NEG_SLOPE 0.2f
#define CDIV(a, b) (((a) + (b) - 1) / (b))

#define NU 100000
#define NP 50000
#define NC 500
#define ESIM 1000000
#define EPREF 1000000
#define EBEL 50000

typedef __attribute__((ext_vector_type(16))) __bf16 v16bf;
typedef __attribute__((ext_vector_type(16))) unsigned short v16us;
typedef __attribute__((ext_vector_type(8))) float v8f;

// ---------------------------------------------------------------- helpers ---
__device__ __forceinline__ unsigned short f2bf(float f) {
    unsigned int u = __float_as_uint(f);
    u += 0x7fffu + ((u >> 16) & 1u);   // round-to-nearest-even
    return (unsigned short)(u >> 16);
}

__device__ __forceinline__ void atomicMaxF(float* addr, float val) {
    unsigned int* ua = (unsigned int*)addr;
    unsigned int cur = *ua;
    while (__uint_as_float(cur) < val) {
        unsigned int prev = atomicCAS(ua, cur, __float_as_uint(val));
        if (prev == cur) break;
        cur = prev;
    }
}

// ---------------------------------------------------------------- kernels ---

// h = relu(x[:, :4] @ W[4,64] + b); also emit bf16 copy
__global__ void k_proj(const float* __restrict__ x, const float* __restrict__ W,
                       const float* __restrict__ b, float* __restrict__ h,
                       unsigned short* __restrict__ hb, int n) {
    __shared__ float Wl[4 * 64];
    __shared__ float bl[64];
    if (threadIdx.x < 256) Wl[threadIdx.x] = W[threadIdx.x];
    if (threadIdx.x < 64) bl[threadIdx.x] = b[threadIdx.x];
    __syncthreads();
    int idx = blockIdx.x * blockDim.x + threadIdx.x;
    if (idx >= n * 64) return;
    int node = idx >> 6, c = idx & 63;
    const float* xr = x + (size_t)node * 4;
    float acc = bl[c] + xr[0] * Wl[c] + xr[1] * Wl[64 + c] +
                xr[2] * Wl[128 + c] + xr[3] * Wl[192 + c];
    acc = fmaxf(acc, 0.f);
    h[idx] = acc;
    hb[idx] = f2bf(acc);
}

__global__ void k_fill(float* __restrict__ p, int n, float v) {
    int idx = blockIdx.x * blockDim.x + threadIdx.x;
    if (idx < n) p[idx] = v;
}

// Pack f32 weight W[K,N] into bf16 WMMA B-fragment order:
// Bp[((col/16)*(K/32) + k/32)*32 + lane][j], lane = (kr/16)*16 + col%16,
// j = kr%16, so each lane's 16 elements are contiguous (one 32B load).
__global__ void k_pack_B(const float* __restrict__ W, unsigned short* __restrict__ Bp,
                         int N, int K) {
    int idx = blockIdx.x * blockDim.x + threadIdx.x;
    if (idx >= N * K) return;
    int k = idx / N, col = idx - k * N;
    int ks = k >> 5, kr = k & 31;
    int half = kr >> 4;
    int lane = half * 16 + (col & 15);
    int j = kr & 15;
    size_t o = ((size_t)((col >> 4) * (K >> 5) + ks) * 32 + lane) * 16 + j;
    Bp[o] = f2bf(W[idx]);
}

// C[M,N] = A[M,KT] @ B[KT,N] (+bias[col]); A row-major bf16, B pre-packed.
// One 16x16 tile per wave, 8 waves/block. N multiple of 16.
template <int KT>
__global__ void k_gemm_bf16(const unsigned short* __restrict__ A,
                            const unsigned short* __restrict__ Bp,
                            float* __restrict__ C, unsigned short* __restrict__ Cbf,
                            const float* __restrict__ bias, int M, int N) {
    constexpr int KSTEPS = KT / 32;
    int lane = threadIdx.x & 31;
    int wave = threadIdx.x >> 5;
    int n_tile = blockIdx.x;
    int m_tile = blockIdx.y * (blockDim.x >> 5) + wave;
    if (m_tile * 16 >= M) return;                 // wave-uniform exit
    int half = lane >> 4;
    int col = n_tile * 16 + (lane & 15);
    int rowA = m_tile * 16 + (lane & 15);
    if (rowA >= M) rowA = M - 1;                  // branchless clamp (rows >= M never stored)
    const uint4* Arow = (const uint4*)(A + (size_t)rowA * KT);
    const unsigned short* Bbase =
        Bp + ((size_t)(n_tile * KSTEPS) * 32 + lane) * 16;
    v8f acc = {};
#pragma unroll
    for (int ks = 0; ks < KSTEPS; ++ks) {
        union { uint4 q[2]; v16us v; } af;
        af.q[0] = Arow[ks * 4 + 2 * half];        // K = ks*32 + half*16 .. +7
        af.q[1] = Arow[ks * 4 + 2 * half + 1];    // K = ks*32 + half*16 + 8 .. +15
        v16us bu = *(const v16us*)(Bbase + (size_t)ks * 32 * 16);
        v16bf a = __builtin_bit_cast(v16bf, af.v);
        v16bf b = __builtin_bit_cast(v16bf, bu);
        acc = __builtin_amdgcn_wmma_f32_16x16x32_bf16(
            false, a, false, b, (short)0, acc, false, false);
    }
    float bb = bias ? bias[col] : 0.f;
#pragma unroll
    for (int r = 0; r < 8; ++r) {
        int row = m_tile * 16 + r + (half ? 8 : 0);
        if (row < M) {
            float v = acc[r] + bb;
            if (C)   C[(size_t)row * N + col] = v;
            if (Cbf) Cbf[(size_t)row * N + col] = f2bf(v);
        }
    }
}

// wvec[k*H+h] = sum_c W[k, h*C+c] * att[h, c]   (fold att into W for logits)
__global__ void k_att_reduce(const float* __restrict__ W, const float* __restrict__ att,
                             float* __restrict__ wvec, int H, int C) {
    int idx = blockIdx.x * blockDim.x + threadIdx.x;
    if (idx >= 64 * H) return;
    int k = idx / H, h = idx % H;
    float acc = 0.f;
    for (int c = 0; c < C; ++c) acc += W[(size_t)k * H * C + h * C + c] * att[h * C + c];
    wvec[idx] = acc;
}

// a[n*H+h] = sum_k h[n,k] * wvec[k*H+h]   (K fixed at 64)
__global__ void k_nodedot(const float* __restrict__ h, const float* __restrict__ wvec,
                          float* __restrict__ a, int n, int H) {
    int idx = blockIdx.x * blockDim.x + threadIdx.x;
    if (idx >= n * H) return;
    int node = idx / H, hh = idx % H;
    const float* hr = h + (size_t)node * 64;
    float acc = 0.f;
    for (int k = 0; k < 64; ++k) acc += hr[k] * wvec[k * H + hh];
    a[idx] = acc;
}

__global__ void k_edge_max(const int* __restrict__ src, const int* __restrict__ dst,
                           int E, int nloop, const float* __restrict__ a_s,
                           const float* __restrict__ a_d, int H, float* __restrict__ m) {
    int idx = blockIdx.x * blockDim.x + threadIdx.x;
    if (idx >= (E + nloop) * H) return;
    int e = idx / H, h = idx - e * H;
    int s, d;
    if (e < E) { s = src[e]; d = dst[e]; } else { s = d = e - E; }
    float al = a_s[(size_t)s * H + h] + a_d[(size_t)d * H + h];
    al = al > 0.f ? al : al * NEG_SLOPE;
    atomicMaxF(&m[(size_t)d * H + h], al);
}

__global__ void k_edge_exp(const int* __restrict__ src, const int* __restrict__ dst,
                           int E, int nloop, const float* __restrict__ a_s,
                           const float* __restrict__ a_d, int H,
                           const float* __restrict__ m, float* __restrict__ ebuf,
                           float* __restrict__ ssum) {
    int idx = blockIdx.x * blockDim.x + threadIdx.x;
    if (idx >= (E + nloop) * H) return;
    int e = idx / H, h = idx - e * H;
    int s, d;
    if (e < E) { s = src[e]; d = dst[e]; } else { s = d = e - E; }
    float al = a_s[(size_t)s * H + h] + a_d[(size_t)d * H + h];
    al = al > 0.f ? al : al * NEG_SLOPE;
    float ev = expf(al - m[(size_t)d * H + h]);
    ebuf[idx] = ev;
    atomicAdd(&ssum[(size_t)d * H + h], ev);
}

// out[dst, h, c0..c0+3] += hs[src, h, c0..c0+3] * (e / s[dst,h]); H*C/4 threads/edge
__global__ void k_edge_scatter(const int* __restrict__ src, const int* __restrict__ dst,
                               int E, int nloop, const float* __restrict__ hs,
                               const float* __restrict__ ebuf,
                               const float* __restrict__ ssum,
                               float* __restrict__ out, int H, int C) {
    int tpe = (H * C) >> 2;
    long long idx = (long long)blockIdx.x * blockDim.x + threadIdx.x;
    long long tot = (long long)(E + nloop) * tpe;
    if (idx >= tot) return;
    int e = (int)(idx / tpe);
    int t = (int)(idx - (long long)e * tpe);
    int h = (t * 4) / C, c = (t * 4) % C;
    int s, d;
    if (e < E) { s = src[e]; d = dst[e]; } else { s = d = e - E; }
    float w = ebuf[(size_t)e * H + h] / ssum[(size_t)d * H + h];
    const float4 v = *(const float4*)(hs + (size_t)s * H * C + h * C + c);
    float* ob = out + (size_t)d * H * C + h * C + c;
    atomicAdd(ob + 0, v.x * w);
    atomicAdd(ob + 1, v.y * w);
    atomicAdd(ob + 2, v.z * w);
    atomicAdd(ob + 3, v.w * w);
}

// out = [relu]( scale * (mean_heads(o) + b1 + b2) ); optional f32 + bf16 outputs
__global__ void k_combine(const float* __restrict__ o, int n, int H, int C,
                          const float* __restrict__ b1, const float* __restrict__ b2,
                          float scale, int do_relu, float* __restrict__ outf,
                          unsigned short* __restrict__ outb) {
    int idx = blockIdx.x * blockDim.x + threadIdx.x;
    if (idx >= n * C) return;
    int node = idx / C, c = idx % C;
    float acc = 0.f;
    for (int h = 0; h < H; ++h) acc += o[(size_t)node * H * C + h * C + c];
    acc = acc / (float)H + b1[c] + (b2 ? b2[c] : 0.f);
    acc *= scale;
    if (do_relu) acc = fmaxf(acc, 0.f);
    if (outf) outf[idx] = acc;
    if (outb) outb[idx] = f2bf(acc);
}

// ------------------------------------------------------------- host driver ---
static void run_gat_relation(hipStream_t stream, const unsigned short* A_bf,
                             const float* hsrc_f, const float* hdst_f, int Nsrc,
                             int Ndst, const int* src, const int* dst, int E,
                             int nloop, const float* W_f, const unsigned short* W_pk,
                             const float* att_s, const float* att_d, int H, int C,
                             float* hs, float* a_s, float* a_d, float* mbuf,
                             float* sbuf, float* ebuf, float* wvs, float* wvd,
                             float* o_acc) {
    int N = H * C;
    k_att_reduce<<<CDIV(64 * H, 64), 64, 0, stream>>>(W_f, att_s, wvs, H, C);
    k_att_reduce<<<CDIV(64 * H, 64), 64, 0, stream>>>(W_f, att_d, wvd, H, C);
    k_nodedot<<<CDIV(Nsrc * H, 256), 256, 0, stream>>>(hsrc_f, wvs, a_s, Nsrc, H);
    k_nodedot<<<CDIV(Ndst * H, 256), 256, 0, stream>>>(hdst_f, wvd, a_d, Ndst, H);
    dim3 g(N / 16, CDIV(CDIV(Nsrc, 16), 8));
    k_gemm_bf16<64><<<g, 256, 0, stream>>>(A_bf, W_pk, hs, (unsigned short*)nullptr,
                                           (const float*)nullptr, Nsrc, N);
    k_fill<<<CDIV(Ndst * H, 256), 256, 0, stream>>>(mbuf, Ndst * H, -1e30f);
    k_fill<<<CDIV(Ndst * H, 256), 256, 0, stream>>>(sbuf, Ndst * H, 0.f);
    int Etot = E + nloop;
    k_edge_max<<<CDIV(Etot * H, 256), 256, 0, stream>>>(src, dst, E, nloop, a_s,
                                                        a_d, H, mbuf);
    k_edge_exp<<<CDIV(Etot * H, 256), 256, 0, stream>>>(src, dst, E, nloop, a_s,
                                                        a_d, H, mbuf, ebuf, sbuf);
    long long tot = (long long)Etot * (N / 4);
    k_edge_scatter<<<(unsigned)CDIV(tot, 256), 256, 0, stream>>>(
        src, dst, E, nloop, hs, ebuf, sbuf, o_acc, H, C);
}

extern "C" void kernel_launch(void* const* d_in, const int* in_sizes, int n_in,
                              void* d_out, int out_size, void* d_ws, size_t ws_size,
                              hipStream_t stream) {
    (void)in_sizes; (void)n_in; (void)out_size; (void)ws_size;
    // ---- inputs (jax pytree order: dict keys sorted) ----
    const float* x_user = (const float*)d_in[0];
    const float* x_prod = (const float*)d_in[1];
    const float* x_cat  = (const float*)d_in[2];
    const int* sim_src  = (const int*)d_in[3];
    const int* sim_dst  = (const int*)d_in[4];
    const int* pref_src = (const int*)d_in[5];
    const int* pref_dst = (const int*)d_in[6];
    const int* bel_src  = (const int*)d_in[7];
    const int* bel_dst  = (const int*)d_in[8];
    // params: c1.{bel,pref,sim}=(W,att_s,att_d,b) -> 9..20
    const float* c1b_W = (const float*)d_in[9],  *c1b_as = (const float*)d_in[10],
               * c1b_ad = (const float*)d_in[11], *c1b_b = (const float*)d_in[12];
    const float* c1p_W = (const float*)d_in[13], *c1p_as = (const float*)d_in[14],
               * c1p_ad = (const float*)d_in[15], *c1p_b = (const float*)d_in[16];
    const float* c1s_W = (const float*)d_in[17], *c1s_as = (const float*)d_in[18],
               * c1s_ad = (const float*)d_in[19], *c1s_b = (const float*)d_in[20];
    // c2.{bel,pref,sim} -> 21..32
    const float* c2b_W = (const float*)d_in[21], *c2b_as = (const float*)d_in[22],
               * c2b_ad = (const float*)d_in[23], *c2b_b = (const float*)d_in[24];
    const float* c2p_W = (const float*)d_in[25], *c2p_as = (const float*)d_in[26],
               * c2p_ad = (const float*)d_in[27], *c2p_b = (const float*)d_in[28];
    const float* c2s_W = (const float*)d_in[29], *c2s_as = (const float*)d_in[30],
               * c2s_ad = (const float*)d_in[31], *c2s_b = (const float*)d_in[32];
    const float* dim_W = (const float*)d_in[33], *dim_b = (const float*)d_in[34];
    const float* oc_W = (const float*)d_in[35], *oc_b = (const float*)d_in[36];
    const float* op_W = (const float*)d_in[37], *op_b = (const float*)d_in[38];
    const float* ou_W = (const float*)d_in[39], *ou_b = (const float*)d_in[40];
    const float* pc_W = (const float*)d_in[41], *pc_b = (const float*)d_in[42];
    const float* pp_W = (const float*)d_in[43], *pp_b = (const float*)d_in[44];
    const float* pu_W = (const float*)d_in[45], *pu_b = (const float*)d_in[46];

    // ---- workspace carve ----
    char* ws = (char*)d_ws;
    size_t off = 0;
    auto alloc = [&](size_t bytes) -> void* {
        void* p = ws + off;
        off = (off + bytes + 255) & ~(size_t)255;
        return p;
    };
    float* h_u_f = (float*)alloc((size_t)NU * 64 * 4);
    float* h_p_f = (float*)alloc((size_t)NP * 64 * 4);
    float* h_c_f = (float*)alloc((size_t)NC * 64 * 4);
    unsigned short* h_u_b = (unsigned short*)alloc((size_t)NU * 64 * 2);
    unsigned short* h_p_b = (unsigned short*)alloc((size_t)NP * 64 * 2);
    unsigned short* h_c_b = (unsigned short*)alloc((size_t)NC * 64 * 2);
    float* hs   = (float*)alloc((size_t)NU * 256 * 4);   // src transform (reused)
    float* o_p  = (float*)alloc((size_t)NP * 256 * 4);   // product accumulator
    float* o_c  = (float*)alloc((size_t)NC * 256 * 4);   // category accumulator
    float* a_s  = (float*)alloc((size_t)NU * 4 * 4);
    float* a_d  = (float*)alloc((size_t)NP * 4 * 4);
    float* mbuf = (float*)alloc((size_t)NP * 4 * 4);
    float* sbuf = (float*)alloc((size_t)NP * 4 * 4);
    float* ebuf = (float*)alloc((size_t)(ESIM + NP) * 4 * 4);
    float* wvs  = (float*)alloc(64 * 4 * 4);
    float* wvd  = (float*)alloc(64 * 4 * 4);
    unsigned short* b_c1s = (unsigned short*)alloc(16384 * 2);  // packed weights
    unsigned short* b_c1p = (unsigned short*)alloc(16384 * 2);
    unsigned short* b_c1b = (unsigned short*)alloc(16384 * 2);
    unsigned short* b_c2s = (unsigned short*)alloc(2048 * 2);
    unsigned short* b_c2p = (unsigned short*)alloc(2048 * 2);
    unsigned short* b_c2b = (unsigned short*)alloc(2048 * 2);
    unsigned short* b_dim = (unsigned short*)alloc(2048 * 2);
    unsigned short* b_ou  = (unsigned short*)alloc(1024 * 2);
    unsigned short* b_op  = (unsigned short*)alloc(1024 * 2);
    unsigned short* b_oc  = (unsigned short*)alloc(1024 * 2);
    unsigned short* h_uf_b = (unsigned short*)alloc((size_t)NU * 32 * 2);
    unsigned short* h_pf_b = (unsigned short*)alloc((size_t)NP * 32 * 2);
    unsigned short* h_cf_b = (unsigned short*)alloc((size_t)NC * 32 * 2);

    // ---- weight packing (f32 -> bf16 WMMA fragment order) ----
    struct { const float* s; unsigned short* d; int N; int K; } pk[10] = {
        {c1s_W, b_c1s, 256, 64}, {c1p_W, b_c1p, 256, 64}, {c1b_W, b_c1b, 256, 64},
        {c2s_W, b_c2s, 32, 64},  {c2p_W, b_c2p, 32, 64},  {c2b_W, b_c2b, 32, 64},
        {dim_W, b_dim, 32, 64},  {ou_W, b_ou, 32, 32},    {op_W, b_op, 32, 32},
        {oc_W, b_oc, 32, 32}};
    for (int i = 0; i < 10; ++i)
        k_pack_B<<<CDIV(pk[i].N * pk[i].K, 256), 256, 0, stream>>>(pk[i].s, pk[i].d,
                                                                   pk[i].N, pk[i].K);

    // ---- projections (4 -> 64, relu) ----
    k_proj<<<CDIV(NU * 64, 256), 256, 0, stream>>>(x_user, pu_W, pu_b, h_u_f, h_u_b, NU);
    k_proj<<<CDIV(NP * 64, 256), 256, 0, stream>>>(x_prod, pp_W, pp_b, h_p_f, h_p_b, NP);
    k_proj<<<CDIV(NC * 64, 256), 256, 0, stream>>>(x_cat, pc_W, pc_b, h_c_f, h_c_b, NC);

    // ---- conv1 (heads=4, C=64) ----
    k_fill<<<CDIV(NP * 256, 256), 256, 0, stream>>>(o_p, NP * 256, 0.f);
    k_fill<<<CDIV(NC * 256, 256), 256, 0, stream>>>(o_c, NC * 256, 0.f);
    run_gat_relation(stream, h_p_b, h_p_f, h_p_f, NP, NP, sim_src, sim_dst, ESIM,
                     NP, c1s_W, b_c1s, c1s_as, c1s_ad, 4, 64, hs, a_s, a_d, mbuf,
                     sbuf, ebuf, wvs, wvd, o_p);
    run_gat_relation(stream, h_u_b, h_u_f, h_p_f, NU, NP, pref_src, pref_dst,
                     EPREF, 0, c1p_W, b_c1p, c1p_as, c1p_ad, 4, 64, hs, a_s, a_d,
                     mbuf, sbuf, ebuf, wvs, wvd, o_p);
    run_gat_relation(stream, h_p_b, h_p_f, h_c_f, NP, NC, bel_src, bel_dst, EBEL,
                     0, c1b_W, b_c1b, c1b_as, c1b_ad, 4, 64, hs, a_s, a_d, mbuf,
                     sbuf, ebuf, wvs, wvd, o_c);
    // h_p = relu(0.5*(mean_h + b_sim + b_pref)); h_c = relu(mean_h + b_bel)
    k_combine<<<CDIV(NP * 64, 256), 256, 0, stream>>>(o_p, NP, 4, 64, c1s_b, c1p_b,
                                                      0.5f, 1, h_p_f, h_p_b);
    k_combine<<<CDIV(NC * 64, 256), 256, 0, stream>>>(o_c, NC, 4, 64, c1b_b,
                                                      (const float*)nullptr, 1.f, 1,
                                                      h_c_f, h_c_b);

    // ---- conv2 (heads=1, C=32) ----
    k_fill<<<CDIV(NP * 32, 256), 256, 0, stream>>>(o_p, NP * 32, 0.f);
    k_fill<<<CDIV(NC * 32, 256), 256, 0, stream>>>(o_c, NC * 32, 0.f);
    run_gat_relation(stream, h_p_b, h_p_f, h_p_f, NP, NP, sim_src, sim_dst, ESIM,
                     NP, c2s_W, b_c2s, c2s_as, c2s_ad, 1, 32, hs, a_s, a_d, mbuf,
                     sbuf, ebuf, wvs, wvd, o_p);
    run_gat_relation(stream, h_u_b, h_u_f, h_p_f, NU, NP, pref_src, pref_dst,
                     EPREF, 0, c2p_W, b_c2p, c2p_as, c2p_ad, 1, 32, hs, a_s, a_d,
                     mbuf, sbuf, ebuf, wvs, wvd, o_p);
    run_gat_relation(stream, h_p_b, h_p_f, h_c_f, NP, NC, bel_src, bel_dst, EBEL,
                     0, c2b_W, b_c2b, c2b_as, c2b_ad, 1, 32, hs, a_s, a_d, mbuf,
                     sbuf, ebuf, wvs, wvd, o_c);
    // h_p = 0.5*(o + b2s + b2p); h_c = o + b2b  (no relu) -> bf16 for out heads
    k_combine<<<CDIV(NP * 32, 256), 256, 0, stream>>>(o_p, NP, 1, 32, c2s_b, c2p_b,
                                                      0.5f, 0, (float*)nullptr, h_pf_b);
    k_combine<<<CDIV(NC * 32, 256), 256, 0, stream>>>(o_c, NC, 1, 32, c2b_b,
                                                      (const float*)nullptr, 1.f, 0,
                                                      (float*)nullptr, h_cf_b);

    // ---- user path: dimproj 64->32 (+bias), bf16 out for final head ----
    {
        dim3 g(32 / 16, CDIV(CDIV(NU, 16), 8));
        k_gemm_bf16<64><<<g, 256, 0, stream>>>(h_u_b, b_dim, (float*)nullptr, h_uf_b,
                                               dim_b, NU, 32);
    }

    // ---- output heads 32->32 (+bias) directly into d_out ----
    float* out_u = (float*)d_out;
    float* out_p = out_u + (size_t)NU * 32;
    float* out_c = out_p + (size_t)NP * 32;
    {
        dim3 g(32 / 16, CDIV(CDIV(NU, 16), 8));
        k_gemm_bf16<32><<<g, 256, 0, stream>>>(h_uf_b, b_ou, out_u,
                                               (unsigned short*)nullptr, ou_b, NU, 32);
    }
    {
        dim3 g(32 / 16, CDIV(CDIV(NP, 16), 8));
        k_gemm_bf16<32><<<g, 256, 0, stream>>>(h_pf_b, b_op, out_p,
                                               (unsigned short*)nullptr, op_b, NP, 32);
    }
    {
        dim3 g(32 / 16, CDIV(CDIV(NC, 16), 8));
        k_gemm_bf16<32><<<g, 256, 0, stream>>>(h_cf_b, b_oc, out_c,
                                               (unsigned short*)nullptr, oc_b, NC, 32);
    }
}
